// AdaptiveAttention_66632122630634
// MI455X (gfx1250) — matrix operand order repaired and checked by reading, not compiled, not run on hardware
//
#include <hip/hip_runtime.h>
#include <math.h>

// ---------------------------------------------------------------- constants
#define B_   2
#define C_   256
#define H_   64
#define W_   64
#define N_   4096          // H_*W_
#define NH_  4
#define HD_  64
#define K_   16
#define INV_SCALE 0.125f   // 1/sqrt(64)

// ---------------------------------------------------------------- WMMA types
typedef __attribute__((ext_vector_type(16))) __bf16 bf16x16;
typedef __attribute__((ext_vector_type(8)))  float  f32x8;

struct alignas(16) B128 { unsigned int x[4]; };          // POD 16-byte chunk
struct alignas(16) F4   { float x[4]; };                 // POD float4
union FragU { bf16x16 v; B128 b[2]; };                   // 32B operand frag

__device__ __forceinline__ f32x8 wmma_bf16(bf16x16 a, bf16x16 b, f32x8 c) {
    return __builtin_amdgcn_wmma_f32_16x16x32_bf16(
        /*neg_a=*/false, a, /*neg_b=*/false, b,
        /*c_mod=*/(short)0, c, /*reuse_a=*/false, /*reuse_b=*/false);
}

// A-operand (16x32 bf16): lane<16 holds K {0..7,16..23}, lane>=16 {8..15,24..31}
__device__ __forceinline__ bf16x16 load_fragA(const __bf16* rowp, int lane) {
    int k0 = (lane & 16) ? 8 : 0;
    FragU u;
    u.b[0] = *(const B128*)(rowp + k0);
    u.b[1] = *(const B128*)(rowp + k0 + 16);
    return u.v;
}
// B-operand (32x16 bf16) from row-major [n][k] LDS: lane<16 K {0..7,16..23}, lane>=16 {8..15,24..31}
__device__ __forceinline__ bf16x16 load_fragB(const __bf16* rowp, int lane) {
    int k0 = (lane & 16) ? 8 : 0;
    FragU u;
    u.b[0] = *(const B128*)(rowp + k0);
    u.b[1] = *(const B128*)(rowp + k0 + 16);
    return u.v;
}

// ---------------------------------------------------------------- zero
__global__ void zero_f32(float* p, int n) {
    int i = blockIdx.x * 256 + threadIdx.x;
    if (i < n) p[i] = 0.f;
}

// ---------------------------------------------------------------- generic WMMA GEMM
// Out[b] = act(A[b] @ W + bias) (+Res[b]); A optionally transposed (A[m,k]=Ap[k*lda+m]).
// 128x64 tile per WG, 128 threads (4 waves); each wave: 32 rows -> 2 A frags,
// 4 shared B frags, 8 back-to-back WMMAs per 32-deep k-step. f32 accumulate.
#define GS 40   // padded LDS row stride (halfs); 80B -> conflict-free frag loads
__global__ __launch_bounds__(128)
void wmma_gemm(const float* __restrict__ A, long long sAb, int aT, int lda,
               const float* __restrict__ Wt, const float* __restrict__ bias,
               const float* __restrict__ Res, long long sRb,
               float* __restrict__ OutF, long long sOFb, int outTrans,
               __bf16* __restrict__ OutB, long long sOBb,
               int M, int Nn, int Kk, int leaky)
{
    __shared__ __bf16 As [128 * GS];   // 10 KB
    __shared__ __bf16 Bts[ 64 * GS];   //  5 KB

    const int tid  = threadIdx.x;
    const int lane = tid & 31;
    const int w    = tid >> 5;
    const int m0   = blockIdx.x * 128;
    const int n0   = blockIdx.y * 64;
    const int b    = blockIdx.z;
    const float* Ab = A + (size_t)b * sAb;

    f32x8 acc[2][4];
    #pragma unroll
    for (int mb = 0; mb < 2; ++mb)
        #pragma unroll
        for (int cb = 0; cb < 4; ++cb)
            #pragma unroll
            for (int r = 0; r < 8; ++r) acc[mb][cb][r] = 0.f;

    const int mr = lane & 15;
    for (int k0 = 0; k0 < Kk; k0 += 32) {
        __syncthreads();
        // ---- stage A tile 128x32 (f32 -> bf16), b128 global loads
        if (!aT) {
            for (int idx = tid; idx < 1024; idx += 128) {
                const int row = idx >> 3;            // 0..127
                const int kk  = (idx & 7) * 4;       // 0,4,..,28
                F4 f = *(const F4*)&Ab[(size_t)(m0 + row) * lda + k0 + kk];
                #pragma unroll
                for (int e = 0; e < 4; ++e)
                    As[row * GS + kk + e] = (__bf16)f.x[e];
            }
        } else {
            for (int idx = tid; idx < 1024; idx += 128) {
                const int row4 = (idx & 31) * 4;     // 0..124 (vector along m)
                const int kk   = idx >> 5;           // 0..31
                F4 f = *(const F4*)&Ab[(size_t)(k0 + kk) * lda + m0 + row4];
                #pragma unroll
                for (int e = 0; e < 4; ++e)
                    As[(row4 + e) * GS + kk] = (__bf16)f.x[e];
            }
        }
        // ---- stage W tile 32x64 transposed -> Bts[n][k]
        for (int idx = tid; idx < 512; idx += 128) {
            const int col4 = (idx & 15) * 4;         // 0..60
            const int kk   = idx >> 4;               // 0..31
            F4 f = *(const F4*)&Wt[(size_t)(k0 + kk) * Nn + n0 + col4];
            #pragma unroll
            for (int e = 0; e < 4; ++e)
                Bts[(col4 + e) * GS + kk] = (__bf16)f.x[e];
        }
        __syncthreads();

        // ---- fragments first, then 8 back-to-back WMMAs
        bf16x16 a0 = load_fragA(&As[(w * 32 + mr) * GS], lane);
        bf16x16 a1 = load_fragA(&As[(w * 32 + 16 + mr) * GS], lane);
        bf16x16 bf[4];
        #pragma unroll
        for (int cb = 0; cb < 4; ++cb)
            bf[cb] = load_fragB(&Bts[(cb * 16 + mr) * GS], lane);
        #pragma unroll
        for (int cb = 0; cb < 4; ++cb)
            acc[0][cb] = wmma_bf16(a0, bf[cb], acc[0][cb]);
        #pragma unroll
        for (int cb = 0; cb < 4; ++cb)
            acc[1][cb] = wmma_bf16(a1, bf[cb], acc[1][cb]);
    }

    // epilogue: C layout -> (m = r + 8*(lane>>4), n = lane&15)
    #pragma unroll
    for (int mb = 0; mb < 2; ++mb) {
        const int rowb = m0 + w * 32 + mb * 16 + ((lane >> 4) << 3);
        #pragma unroll
        for (int cb = 0; cb < 4; ++cb) {
            const int col = n0 + cb * 16 + mr;
            #pragma unroll
            for (int r = 0; r < 8; ++r) {
                const int row = rowb + r;
                float val = acc[mb][cb][r] + bias[col];
                if (leaky) val = val > 0.f ? val : 0.01f * val;
                if (Res)  val += Res[(size_t)b * sRb + (size_t)row * Nn + col];
                if (OutF) {
                    size_t oi = outTrans ? ((size_t)col * M + row)
                                         : ((size_t)row * Nn + col);
                    OutF[(size_t)b * sOFb + oi] = val;
                }
                if (OutB)
                    OutB[(size_t)b * sOBb + (size_t)row * Nn + col] = (__bf16)val;
            }
        }
    }
}

// ---------------------------------------------------------------- cluster centers
// Segmented sum of k over labels. thread t owns channel t -> race-free LDS.
__global__ __launch_bounds__(256)
void centers_accum(const __bf16* __restrict__ kb, const int* __restrict__ labels,
                   float* __restrict__ cnum, float* __restrict__ ccnt)
{
    __shared__ float part[K_ * C_];   // 16 KB
    __shared__ float cntl[K_];
    const int b = blockIdx.y, n0 = blockIdx.x * 256, t = threadIdx.x;
    #pragma unroll
    for (int kc = 0; kc < K_; ++kc) part[kc * C_ + t] = 0.f;
    if (t < K_) cntl[t] = 0.f;
    __syncthreads();
    for (int n = 0; n < 256; ++n) {
        const int lab = labels[b * N_ + n0 + n];
        part[lab * C_ + t] += (float)kb[((size_t)(b * N_ + n0 + n)) * C_ + t];
        if (t == 0) cntl[lab] += 1.f;
    }
    __syncthreads();
    #pragma unroll
    for (int kc = 0; kc < K_; ++kc)
        atomicAdd(&cnum[((size_t)(b * K_) + kc) * C_ + t], part[kc * C_ + t]);
    if (t < K_) atomicAdd(&ccnt[b * K_ + t], cntl[t]);
}

__global__ void centers_fin(const float* cnum, const float* ccnt, float* centers) {
    int i = blockIdx.x * 256 + threadIdx.x;
    if (i < B_ * K_ * C_) centers[i] = cnum[i] / (ccnt[i / C_] + 1e-6f);
}

// ---------------------------------------------------------------- attn_centers: ac[b,h,n,kc] = q . c
__global__ __launch_bounds__(256)
void ac_kernel(const __bf16* __restrict__ qb, const float* __restrict__ centers,
               float* __restrict__ ac)
{
    __shared__ float cs[K_ * HD_];    // 4 KB
    const int bh = blockIdx.y, b = bh / NH_, h = bh % NH_;
    const int tid = threadIdx.x;
    for (int idx = tid; idx < K_ * HD_; idx += 256) {
        int kc = idx >> 6, d = idx & 63;
        cs[idx] = centers[((size_t)(b * K_) + kc) * C_ + h * HD_ + d];
    }
    __syncthreads();
    const int n = blockIdx.x * 256 + tid;
    const __bf16* qp = qb + ((size_t)(b * N_ + n)) * C_ + h * HD_;
    float qr[HD_];
    #pragma unroll
    for (int d = 0; d < HD_; ++d) qr[d] = (float)qp[d];
    #pragma unroll
    for (int kc = 0; kc < K_; ++kc) {
        float s = 0.f;
        #pragma unroll
        for (int d = 0; d < HD_; ++d) s += qr[d] * cs[kc * HD_ + d];
        ac[((size_t)bh * N_ + n) * K_ + kc] = s;
    }
}

// ---------------------------------------------------------------- flash attention
// WG = 64 q-rows x (b,h); 4 waves x 16 rows each; stream 64-key tiles.
#define AS 72   // padded LDS stride (144B) -> conflict-free b128 frag loads
__global__ __launch_bounds__(128)
void attn_kernel(const __bf16* __restrict__ qb, const __bf16* __restrict__ kb,
                 const __bf16* __restrict__ vb, const int* __restrict__ labels,
                 const float* __restrict__ pc, const float* __restrict__ ac,
                 float* __restrict__ outp)
{
    __shared__ __bf16 Ks [64 * AS];
    __shared__ __bf16 Vts[64 * AS];
    __shared__ __bf16 Ps [64 * AS];
    __shared__ int    labj[64];

    const int tid  = threadIdx.x;
    const int lane = tid & 31;
    const int w    = tid >> 5;
    const int b    = blockIdx.z, h = blockIdx.y;
    const int i0   = blockIdx.x * 64;
    const int mr   = lane & 15;
    const int ka0  = (lane & 16) ? 8 : 0;

    // q A-fragments held in registers for the whole kernel
    const int qrow = i0 + w * 16 + mr;
    const __bf16* qp = qb + ((size_t)(b * N_ + qrow)) * C_ + h * HD_;
    FragU qa0, qa1;
    qa0.b[0] = *(const B128*)(qp + ka0);       qa0.b[1] = *(const B128*)(qp + ka0 + 16);
    qa1.b[0] = *(const B128*)(qp + 32 + ka0);  qa1.b[1] = *(const B128*)(qp + 32 + ka0 + 16);

    // labels of the 8 rows this lane owns in C-layout
    const int rbase = i0 + w * 16 + ((lane >> 4) << 3);
    int lab_i[8];
    #pragma unroll
    for (int r = 0; r < 8; ++r) lab_i[r] = labels[b * N_ + rbase + r];

    float mrun[8], lrun[8];
    f32x8 oacc[4];
    #pragma unroll
    for (int r = 0; r < 8; ++r) { mrun[r] = -3.0e38f; lrun[r] = 0.f; }
    #pragma unroll
    for (int db = 0; db < 4; ++db)
        #pragma unroll
        for (int r = 0; r < 8; ++r) oacc[db][r] = 0.f;

    const float* pcb = pc + (size_t)b * N_ * N_;
    const float* acb = ac + ((size_t)(b * NH_ + h)) * N_ * K_;

    for (int j0 = 0; j0 < N_; j0 += 64) {
        __syncthreads();
        // stage K (row-major) and V (transposed) tiles, 8-half chunks
        for (int idx = tid; idx < 512; idx += 128) {
            const int jr = idx >> 3;
            const int dc = (idx & 7) * 8;
            const size_t gro = ((size_t)(b * N_ + j0 + jr)) * C_ + h * HD_ + dc;
            *(B128*)&Ks[jr * AS + dc] = *(const B128*)(kb + gro);
            union { B128 q; __bf16 hh[8]; } uv;
            uv.q = *(const B128*)(vb + gro);
            #pragma unroll
            for (int e = 0; e < 8; ++e) Vts[(dc + e) * AS + jr] = uv.hh[e];
        }
        if (tid < 64) labj[tid] = labels[b * N_ + j0 + tid];
        __syncthreads();

        // S = q @ k^T  (Ks row-major is exactly the B-operand [n][k] layout)
        // process column-blocks in pairs: 4 frag loads -> 4 back-to-back WMMAs
        f32x8 sacc[4];
        #pragma unroll
        for (int pr = 0; pr < 2; ++pr) {
            const int c0 = pr * 2, c1 = pr * 2 + 1;
            const __bf16* kp0 = &Ks[(c0 * 16 + mr) * AS];
            const __bf16* kp1 = &Ks[(c1 * 16 + mr) * AS];
            bf16x16 b00 = load_fragB(kp0,      lane);
            bf16x16 b01 = load_fragB(kp0 + 32, lane);
            bf16x16 b10 = load_fragB(kp1,      lane);
            bf16x16 b11 = load_fragB(kp1 + 32, lane);
            f32x8 ca, cb2;
            #pragma unroll
            for (int r = 0; r < 8; ++r) { ca[r] = 0.f; cb2[r] = 0.f; }
            ca  = wmma_bf16(qa0.v, b00, ca);
            cb2 = wmma_bf16(qa0.v, b10, cb2);
            ca  = wmma_bf16(qa1.v, b01, ca);
            cb2 = wmma_bf16(qa1.v, b11, cb2);
            sacc[c0] = ca;
            sacc[c1] = cb2;
        }

        // per-element intra/inter select + scale
        #pragma unroll
        for (int cb = 0; cb < 4; ++cb) {
            const int nloc = cb * 16 + mr;
            const int jcol = j0 + nloc;
            const int lj   = labj[nloc];
            #pragma unroll
            for (int r = 0; r < 8; ++r) {
                float s = sacc[cb][r];
                if (lab_i[r] != lj) {
                    const int row = rbase + r;
                    s = acb[(size_t)row * K_ + lj] * pcb[(size_t)row * N_ + jcol];
                }
                sacc[cb][r] = s * INV_SCALE;
            }
        }

        // online softmax (row reductions across 16 lanes of each half-wave)
        float mnew[8], alpha[8], rsum[8];
        #pragma unroll
        for (int r = 0; r < 8; ++r) {
            float mx = fmaxf(fmaxf(sacc[0][r], sacc[1][r]),
                             fmaxf(sacc[2][r], sacc[3][r]));
            #pragma unroll
            for (int off = 8; off >= 1; off >>= 1)
                mx = fmaxf(mx, __shfl_xor(mx, off, 32));
            mnew[r]  = fmaxf(mrun[r], mx);
            alpha[r] = __expf(mrun[r] - mnew[r]);
            mrun[r]  = mnew[r];
            lrun[r] *= alpha[r];
            rsum[r]  = 0.f;
        }
        #pragma unroll
        for (int db = 0; db < 4; ++db)
            #pragma unroll
            for (int r = 0; r < 8; ++r) oacc[db][r] *= alpha[r];

        #pragma unroll
        for (int cb = 0; cb < 4; ++cb)
            #pragma unroll
            for (int r = 0; r < 8; ++r) {
                float p = __expf(sacc[cb][r] - mnew[r]);
                rsum[r] += p;
                Ps[(w * 16 + r + ((lane >> 4) << 3)) * AS + cb * 16 + mr] = (__bf16)p;
            }
        #pragma unroll
        for (int r = 0; r < 8; ++r) {
            float rs = rsum[r];
            #pragma unroll
            for (int off = 8; off >= 1; off >>= 1)
                rs += __shfl_xor(rs, off, 32);
            lrun[r] += rs;
        }

        // O += P @ V  (Vts[d][j] is the B-operand layout; Ps rows are A rows)
        const __bf16* prow = &Ps[(w * 16 + mr) * AS];
        FragU pa0, pa1;
        pa0.b[0] = *(const B128*)(prow + ka0);       pa0.b[1] = *(const B128*)(prow + ka0 + 16);
        pa1.b[0] = *(const B128*)(prow + 32 + ka0);  pa1.b[1] = *(const B128*)(prow + 32 + ka0 + 16);
        #pragma unroll
        for (int pr = 0; pr < 2; ++pr) {
            const int d0 = pr * 2, d1 = pr * 2 + 1;
            const __bf16* vp0 = &Vts[(d0 * 16 + mr) * AS];
            const __bf16* vp1 = &Vts[(d1 * 16 + mr) * AS];
            bf16x16 b00 = load_fragB(vp0,      lane);
            bf16x16 b01 = load_fragB(vp0 + 32, lane);
            bf16x16 b10 = load_fragB(vp1,      lane);
            bf16x16 b11 = load_fragB(vp1 + 32, lane);
            oacc[d0] = wmma_bf16(pa0.v, b00, oacc[d0]);
            oacc[d1] = wmma_bf16(pa0.v, b10, oacc[d1]);
            oacc[d0] = wmma_bf16(pa1.v, b01, oacc[d0]);
            oacc[d1] = wmma_bf16(pa1.v, b11, oacc[d1]);
        }
    }

    // normalize + write out[b, n, h*64+d]
    #pragma unroll
    for (int r = 0; r < 8; ++r) {
        const float inv = 1.f / lrun[r];
        const int row = rbase + r;
        #pragma unroll
        for (int db = 0; db < 4; ++db)
            outp[((size_t)(b * N_ + row)) * C_ + h * HD_ + db * 16 + mr] =
                oacc[db][r] * inv;
    }
}

// ---------------------------------------------------------------- launch
extern "C" void kernel_launch(void* const* d_in, const int* in_sizes, int n_in,
                              void* d_out, int out_size, void* d_ws, size_t ws_size,
                              hipStream_t stream)
{
    const float* q_img  = (const float*)d_in[0];
    const float* k_img  = (const float*)d_in[1];
    const float* v_img  = (const float*)d_in[2];
    const int*   labels = (const int*)  d_in[3];
    const float* pc     = (const float*)d_in[4];
    const float* Wq  = (const float*)d_in[5];  const float* bq  = (const float*)d_in[6];
    const float* Wk  = (const float*)d_in[7];  const float* bk  = (const float*)d_in[8];
    const float* Wv  = (const float*)d_in[9];  const float* bv  = (const float*)d_in[10];
    const float* W11 = (const float*)d_in[11]; const float* b11 = (const float*)d_in[12];
    const float* W12 = (const float*)d_in[13]; const float* b12 = (const float*)d_in[14];
    const float* W21 = (const float*)d_in[15]; const float* b21 = (const float*)d_in[16];
    const float* W22 = (const float*)d_in[17]; const float* b22 = (const float*)d_in[18];

    char* ws = (char*)d_ws;
    size_t off = 0;
    auto alloc = [&](size_t bytes) -> void* {
        void* p = ws + off;
        off = (off + bytes + 255) & ~(size_t)255;
        return p;
    };
    __bf16* qb      = (__bf16*)alloc((size_t)B_ * N_ * C_ * 2);
    __bf16* kb      = (__bf16*)alloc((size_t)B_ * N_ * C_ * 2);
    __bf16* vb      = (__bf16*)alloc((size_t)B_ * N_ * C_ * 2);
    float*  vf      = (float*) alloc((size_t)B_ * N_ * C_ * 4);
    float*  cnum    = (float*) alloc((size_t)B_ * K_ * C_ * 4);
    float*  ccnt    = (float*) alloc((size_t)B_ * K_ * 4);
    float*  centers = (float*) alloc((size_t)B_ * K_ * C_ * 4);
    float*  ac      = (float*) alloc((size_t)B_ * NH_ * N_ * K_ * 4);
    float*  attnout = (float*) alloc((size_t)B_ * N_ * C_ * 4);
    float*  h1      = (float*) alloc((size_t)B_ * N_ * 2 * C_ * 4);
    float*  rs1     = (float*) alloc((size_t)B_ * N_ * C_ * 4);
    float*  outimg  = (float*)d_out;

    const long long sImg = (long long)C_ * N_;   // NCHW per-batch stride
    const long long sNC  = (long long)N_ * C_;   // [N,C] per-batch stride
    const long long sNH2 = (long long)N_ * 2 * C_;

    // zero center accumulators
    zero_f32<<<dim3((B_ * K_ * C_ + 255) / 256), 256, 0, stream>>>(cnum, B_ * K_ * C_);
    zero_f32<<<dim3(1), 256, 0, stream>>>(ccnt, B_ * K_);

    // q/k/v projections: x (gathered transposed from NCHW) @ W + b
    dim3 gproj(N_ / 128, C_ / 64, B_);
    wmma_gemm<<<gproj, 128, 0, stream>>>(q_img, sImg, 1, N_, Wq, bq,
        nullptr, 0, nullptr, 0, 0, qb, sNC, N_, C_, C_, 0);
    wmma_gemm<<<gproj, 128, 0, stream>>>(k_img, sImg, 1, N_, Wk, bk,
        nullptr, 0, nullptr, 0, 0, kb, sNC, N_, C_, C_, 0);
    wmma_gemm<<<gproj, 128, 0, stream>>>(v_img, sImg, 1, N_, Wv, bv,
        nullptr, 0, vf, sNC, 0, vb, sNC, N_, C_, C_, 0);

    // cluster centers of k
    centers_accum<<<dim3(N_ / 256, B_), 256, 0, stream>>>(kb, labels, cnum, ccnt);
    centers_fin<<<dim3((B_ * K_ * C_ + 255) / 256), 256, 0, stream>>>(cnum, ccnt, centers);

    // q . center scores
    ac_kernel<<<dim3(N_ / 256, B_ * NH_), 256, 0, stream>>>(qb, centers, ac);

    // masked flash attention
    attn_kernel<<<dim3(N_ / 64, NH_, B_), 128, 0, stream>>>(
        qb, kb, vb, labels, pc, ac, attnout);

    // rs1 = v + MLP1(out);  rs2 = rs1 + MLP2(rs1)  (final GEMM stores [B,C,N])
    dim3 gmlpA(N_ / 128, (2 * C_) / 64, B_);
    dim3 gmlpB(N_ / 128, C_ / 64, B_);
    wmma_gemm<<<gmlpA, 128, 0, stream>>>(attnout, sNC, 0, C_, W11, b11,
        nullptr, 0, h1, sNH2, 0, nullptr, 0, N_, 2 * C_, C_, 1);
    wmma_gemm<<<gmlpB, 128, 0, stream>>>(h1, sNH2, 0, 2 * C_, W12, b12,
        vf, sNC, rs1, sNC, 0, nullptr, 0, N_, C_, 2 * C_, 0);
    wmma_gemm<<<gmlpA, 128, 0, stream>>>(rs1, sNC, 0, C_, W21, b21,
        nullptr, 0, h1, sNH2, 0, nullptr, 0, N_, 2 * C_, C_, 1);
    wmma_gemm<<<gmlpB, 128, 0, stream>>>(h1, sNH2, 0, 2 * C_, W22, b22,
        rs1, sNC, outimg, sNC, 1, nullptr, 0, N_, C_, 2 * C_, 0);
}
